// EdgeConv_4509715661541
// MI455X (gfx1250) — matrix operand order repaired and checked by reading, not compiled, not run on hardware
//
#include <hip/hip_runtime.h>
#include <hip/hip_bf16.h>

// ---------------- problem constants ----------------
#define BATCH   16
#define NPTS    4096
#define KNBR    20
#define HIDCH   64
#define NEGSL   0.2f
#define EPSV    1e-5f

static const long long RTOT = (long long)BATCH * NPTS * KNBR;   // 1,310,720 edge rows

typedef __bf16 bf16;
typedef __attribute__((ext_vector_type(16))) bf16  v16bf;
typedef __attribute__((ext_vector_type(8)))  float v8f;

// =====================================================================
// 1) KNN: per batch, stage all 4096 points (+|x|^2) in LDS, each thread
//    owns one query and keeps a sorted top-20 in registers.
//    Ordering key: 2*x_n.x_m - |x_m|^2  (|x_n|^2 is constant per query).
// =====================================================================
__global__ __launch_bounds__(256) void knn_topk_kernel(
    const float* __restrict__ x, int* __restrict__ nbr)
{
    extern __shared__ float lds[];
    float* pts = lds;              // [NPTS*3]
    float* sq  = lds + NPTS * 3;   // [NPTS]

    const int b = blockIdx.y;
    const float* xb = x + (size_t)b * 3 * NPTS;   // [3][N]

    for (int i = threadIdx.x; i < NPTS; i += blockDim.x) {
        float px = xb[i], py = xb[NPTS + i], pz = xb[2 * NPTS + i];
        pts[3 * i] = px; pts[3 * i + 1] = py; pts[3 * i + 2] = pz;
        sq[i] = px * px + py * py + pz * pz;
    }
    __syncthreads();

    const int n = blockIdx.x * 256 + threadIdx.x;
    const float qx = pts[3 * n], qy = pts[3 * n + 1], qz = pts[3 * n + 2];

    float best[KNBR]; int bidx[KNBR];
#pragma unroll
    for (int j = 0; j < KNBR; ++j) { best[j] = -1e30f; bidx[j] = 0; }

    for (int m = 0; m < NPTS; ++m) {           // uniform m -> LDS broadcast reads
        float s = 2.f * (qx * pts[3 * m] + qy * pts[3 * m + 1] + qz * pts[3 * m + 2]) - sq[m];
        if (s > best[KNBR - 1]) {
            float cs = s; int ci = m;
#pragma unroll
            for (int j = 0; j < KNBR; ++j) {
                if (cs > best[j]) {
                    float tb = best[j]; int ti = bidx[j];
                    best[j] = cs; bidx[j] = ci; cs = tb; ci = ti;
                }
            }
        }
    }
    int* op = nbr + ((size_t)b * NPTS + n) * KNBR;
#pragma unroll
    for (int j = 0; j < KNBR; ++j) op[j] = bidx[j];
}

// =====================================================================
// 2) Fused layer GEMM via v_wmma_f32_16x16x32_bf16.
//    One wave = 16 rows x 64 out-channels (4 N-tiles; 1 or 2 K-steps).
//    Epilogue: LeakyReLU, per-channel sum/sumsq (LDS then global atomics),
//    bf16 store of RAW post-LReLU activations (BN folded into next layer).
// =====================================================================
template<bool LAYER1>
__global__ __launch_bounds__(256) void edgeconv_gemm_kernel(
    const float* __restrict__ x,      // LAYER1: [B,3,N]
    const int*   __restrict__ nbr,    // LAYER1: [B,N,K]
    const bf16*  __restrict__ yin,    // !LAYER1: [R,64] raw prev activations
    const bf16*  __restrict__ Wb,     // [64][64] bf16, norm-folded, zero-padded
    const float* __restrict__ bias,   // [64] folded bias
    bf16*  __restrict__ yout,         // [R,64]
    float* __restrict__ gsum, float* __restrict__ gsq)
{
    __shared__ bf16  sW[64 * 64];        // 8 KB weights
    __shared__ bf16  sA[8 * 16 * 32];    // 8 KB layer-1 A staging (8 waves)
    __shared__ float ssum[64], ssq[64];

    const int tid = threadIdx.x;
    for (int i = tid; i < 64 * 64; i += 256) sW[i] = Wb[i];
    if (tid < 64) { ssum[tid] = 0.f; ssq[tid] = 0.f; }

    const int wave = tid >> 5, lane = tid & 31;   // wave32
    const int rowBase = (blockIdx.x * 8 + wave) * 16;

    if constexpr (LAYER1) {
        bf16* aw = sA + wave * 512;
        if (lane < 16) {                       // one edge row per low lane
            int r  = rowBase + lane;
            int bn = r / KNBR;
            int n  = bn & (NPTS - 1);
            int b  = bn >> 12;
            const float* xb = x + (size_t)b * 3 * NPTS;
            int m = nbr[r];
            float c0 = xb[n],          c1 = xb[NPTS + n],     c2 = xb[2 * NPTS + n];
            float f0 = xb[m] - c0,     f1 = xb[NPTS + m] - c1, f2 = xb[2 * NPTS + m] - c2;
            bf16* rowp = aw + lane * 32;
            rowp[0] = (bf16)f0; rowp[1] = (bf16)f1; rowp[2] = (bf16)f2;
            rowp[3] = (bf16)c0; rowp[4] = (bf16)c1; rowp[5] = (bf16)c2;
#pragma unroll
            for (int c = 6; c < 32; ++c) rowp[c] = (bf16)0.f;   // K padding
        }
    }
    __syncthreads();

    // ---- A fragments: lane holds row (lane&15), K chunks at kb / kb+16 ----
    const int arow = lane & 15;
    const int kb   = (lane >> 4) * 8;
    v16bf a0{}, a1{};
    if constexpr (LAYER1) {
        const bf16* rp = sA + wave * 512 + arow * 32;
#pragma unroll
        for (int i = 0; i < 8; ++i) { a0[i] = rp[kb + i]; a0[8 + i] = rp[kb + 16 + i]; }
    } else {
        const bf16* rp = yin + (size_t)(rowBase + arow) * 64;
#pragma unroll
        for (int i = 0; i < 8; ++i) {
            a0[i]     = rp[kb + i];       a0[8 + i] = rp[kb + 16 + i];
            a1[i]     = rp[32 + kb + i];  a1[8 + i] = rp[32 + kb + 16 + i];
        }
    }

    // ---- B = W^T fragments + WMMA + epilogue, 4 N-tiles ----
    const int col   = lane & 15;
    const int kOff  = (lane >> 4) * 8;
    const int rhalf = (lane >> 4) * 8;
#pragma unroll
    for (int t = 0; t < 4; ++t) {
        const int och = t * 16 + col;
        const bf16* wr = sW + och * 64;       // row of W == column of B
        v16bf b0{}, b1{};
#pragma unroll
        for (int i = 0; i < 16; ++i) {
            b0[i] = wr[kOff + i + (i >= 8 ? 8 : 0)];
            if constexpr (!LAYER1) b1[i] = wr[32 + kOff + i + (i >= 8 ? 8 : 0)];
        }
        float bv = bias[och];
        v8f c;
#pragma unroll
        for (int r = 0; r < 8; ++r) c[r] = bv;

        c = __builtin_amdgcn_wmma_f32_16x16x32_bf16(false, a0, false, b0, (short)0, c, false, false);
        if constexpr (!LAYER1)
            c = __builtin_amdgcn_wmma_f32_16x16x32_bf16(false, a1, false, b1, (short)0, c, false, false);

        float lsum = 0.f, lsq = 0.f;
#pragma unroll
        for (int r = 0; r < 8; ++r) {
            float v = c[r];
            v = v > 0.f ? v : NEGSL * v;                  // LeakyReLU
            lsum += v; lsq += v * v;
            yout[(size_t)(rowBase + rhalf + r) * 64 + och] = (bf16)v;
        }
        atomicAdd(&ssum[och], lsum);
        atomicAdd(&ssq[och], lsq);
    }
    __syncthreads();
    if (tid < 64) { atomicAdd(&gsum[tid], ssum[tid]); atomicAdd(&gsq[tid], ssq[tid]); }
}

// =====================================================================
// 3) Tiny glue kernels
// =====================================================================
__global__ void zero_stats_kernel(float* p, int n)
{
    int i = blockIdx.x * 256 + threadIdx.x;
    if (i < n) p[i] = 0.f;
}

// s_c = gamma*rsqrt(var+eps); t_c = beta - mu*s_c
__global__ void finalize_stats_kernel(
    const float* __restrict__ sum, const float* __restrict__ sq,
    const float* __restrict__ gamma, const float* __restrict__ beta,
    float cnt, float* __restrict__ s, float* __restrict__ t)
{
    int c = threadIdx.x;
    float mu  = sum[c] / cnt;
    float var = sq[c] / cnt - mu * mu;
    float sc  = gamma[c] * rsqrtf(var + EPSV);
    s[c] = sc;
    t[c] = beta[c] - mu * sc;
}

// Fold input-channel affine (s,t) into weights: W'[o,c]=W[o,c]*s_c (bf16),
// bias[o] = sum_c W[o,c]*t_c. Pads in_ch -> 64 with zeros. s/t==null => identity.
__global__ void fold_weights_kernel(
    const float* __restrict__ W, int in_ch,
    const float* __restrict__ s, const float* __restrict__ t,
    bf16* __restrict__ Wb, float* __restrict__ bias)
{
    int o = threadIdx.x;           // 64 threads
    float acc = 0.f;
    for (int c = 0; c < 64; ++c) {
        float w  = (c < in_ch) ? W[o * in_ch + c] : 0.f;
        float sc = s ? s[c] : 1.f;
        float tc = t ? t[c] : 0.f;
        Wb[o * 64 + c] = (bf16)(w * sc);
        acc += w * tc;
    }
    bias[o] = acc;
}

// =====================================================================
// 4) Final BN(3) + max over K + transpose to [B,64,N]
//    Consecutive threads share (b,n) and span channels -> coalesced reads.
// =====================================================================
__global__ __launch_bounds__(256) void max_over_k_kernel(
    const bf16* __restrict__ y3, const float* __restrict__ s,
    const float* __restrict__ t, float* __restrict__ out)
{
    int gid = blockIdx.x * 256 + threadIdx.x;   // over B*N*64
    int o   = gid & 63;
    int bn  = gid >> 6;
    float sc = s[o], tc = t[o];
    const bf16* p = y3 + (size_t)bn * KNBR * 64 + o;
    float m = -1e30f;
#pragma unroll
    for (int k = 0; k < KNBR; ++k) {
        float v = (float)p[k * 64];
        v = v * sc + tc;
        m = fmaxf(m, v);
    }
    int b = bn >> 12, n = bn & (NPTS - 1);
    out[(size_t)b * 64 * NPTS + (size_t)o * NPTS + n] = m;
}

// =====================================================================
// Launcher
// =====================================================================
extern "C" void kernel_launch(void* const* d_in, const int* in_sizes, int n_in,
                              void* d_out, int out_size, void* d_ws, size_t ws_size,
                              hipStream_t stream)
{
    const float* x  = (const float*)d_in[0];
    const float* W1 = (const float*)d_in[1];
    const float* W2 = (const float*)d_in[2];
    const float* W3 = (const float*)d_in[3];
    const float* g1 = (const float*)d_in[4];
    const float* b1 = (const float*)d_in[5];
    const float* g2 = (const float*)d_in[6];
    const float* b2 = (const float*)d_in[7];
    float* out = (float*)d_out;

    // ---- workspace carve-up (~342 MB) ----
    char* ws = (char*)d_ws;
    size_t off = 0;
    auto alloc = [&](size_t bytes) -> void* {
        void* p = ws + off;
        off += (bytes + 255) & ~(size_t)255;
        return p;
    };
    int*   nbr   = (int*)  alloc((size_t)RTOT * sizeof(int));
    bf16*  bufA  = (bf16*) alloc((size_t)RTOT * 64 * sizeof(bf16));
    bf16*  bufB  = (bf16*) alloc((size_t)RTOT * 64 * sizeof(bf16));
    bf16*  Wb    = (bf16*) alloc(3 * 64 * 64 * sizeof(bf16));
    float* bias  = (float*)alloc(3 * 64 * sizeof(float));
    float* stats = (float*)alloc(6 * 64 * sizeof(float));   // sum/sq x3
    float* st    = (float*)alloc(6 * 64 * sizeof(float));   // s/t   x3

    const unsigned gemmBlocks = (unsigned)(RTOT / 128);     // 8 waves x 16 rows
    const float cnt = (float)RTOT;

    zero_stats_kernel<<<2, 256, 0, stream>>>(stats, 6 * 64);

    knn_topk_kernel<<<dim3(NPTS / 256, BATCH), 256, (size_t)NPTS * 4 * sizeof(float), stream>>>(x, nbr);

    // layer 1: features(6) -> 64
    fold_weights_kernel<<<1, 64, 0, stream>>>(W1, 6, nullptr, nullptr, Wb, bias);
    edgeconv_gemm_kernel<true><<<gemmBlocks, 256, 0, stream>>>(
        x, nbr, nullptr, Wb, bias, bufA, stats, stats + 64);
    finalize_stats_kernel<<<1, 64, 0, stream>>>(stats, stats + 64, g1, b1, cnt, st, st + 64);

    // layer 2: 64 -> 64, BN1 folded into W2
    fold_weights_kernel<<<1, 64, 0, stream>>>(W2, 64, st, st + 64, Wb + 4096, bias + 64);
    edgeconv_gemm_kernel<false><<<gemmBlocks, 256, 0, stream>>>(
        nullptr, nullptr, bufA, Wb + 4096, bias + 64, bufB, stats + 128, stats + 192);
    finalize_stats_kernel<<<1, 64, 0, stream>>>(stats + 128, stats + 192, g1, b1, cnt, st + 128, st + 192);

    // layer 3: 64 -> 64, BN2 folded into W3
    fold_weights_kernel<<<1, 64, 0, stream>>>(W3, 64, st + 128, st + 192, Wb + 8192, bias + 128);
    edgeconv_gemm_kernel<false><<<gemmBlocks, 256, 0, stream>>>(
        nullptr, nullptr, bufB, Wb + 8192, bias + 128, bufA, stats + 256, stats + 320);
    finalize_stats_kernel<<<1, 64, 0, stream>>>(stats + 256, stats + 320, g2, b2, cnt, st + 256, st + 320);

    // BN3 + max over K + transpose
    max_over_k_kernel<<<(BATCH * NPTS * 64) / 256, 256, 0, stream>>>(bufA, st + 256, st + 320, out);
}
